// HollowTransformerLayer_60052232733088
// MI455X (gfx1250) — matrix operand order repaired and checked by reading, not compiled, not run on hardware
//
#include <hip/hip_runtime.h>
#include <hip/hip_bf16.h>

// ---------------------------------------------------------------------------
// HollowTransformerLayer on MI455X (gfx1250, wave32, WMMA).
// Diagonal-only attention ==> attention output == V, so Q/K are skipped.
// Pipeline:
//   xb  = bf16(x)
//   vb  = bf16(xb @ Wv + bv)                 [WMMA GEMM, async-LDS pipelined]
//   att = f32 (vb @ Wo + bo)                 [WMMA GEMM]
//   h   = LN(x + att)*g1+b1   (f32 + bf16)   [fused residual LN]
//   f1  = bf16(relu(hb @ W1 + b1))           [WMMA GEMM]
//   f2  = f32 (f1 @ W2 + b2)                 [WMMA GEMM]
//   out = LN(h + f2)*g2+b2    (f32)          [fused residual LN]
// GEMMs use statically double-buffered LDS fed by
// GLOBAL_LOAD_ASYNC_TO_LDS_B128 (ASYNCcnt): copy of tile i+1 overlaps the
// WMMAs of tile i; fragment loads are batched ahead of the WMMA chain so
// DScnt waits stagger instead of draining to zero per WMMA.
// ---------------------------------------------------------------------------

typedef __bf16        v16bf __attribute__((ext_vector_type(16)));
typedef float         v8f   __attribute__((ext_vector_type(8)));
typedef unsigned int  v4u   __attribute__((ext_vector_type(4)));

enum { ROWS = 4096, DMQ = 1024, FFQ = 4096 };  // B*S, d_model, d_ff

__device__ __forceinline__ unsigned short f2bf(float f) {
  // round-to-nearest-even fp32 -> bf16
  unsigned int u = __float_as_uint(f);
  unsigned int r = u + 0x7fffu + ((u >> 16) & 1u);
  return (unsigned short)(r >> 16);
}

// Flat pointers into LDS carry the LDS byte offset in addr[31:0] (ISA 10.2),
// which is exactly the VDST operand of the async-to-LDS instructions.
__device__ __forceinline__ unsigned lds_off_of(const void* p) {
  return (unsigned)(unsigned long long)p;
}

// Direct global -> LDS async copy, 16B per active lane. Tracked by ASYNCcnt.
__device__ __forceinline__ void async_copy_b128(unsigned lds_off,
                                                const void* gptr) {
  asm volatile("global_load_async_to_lds_b128 %0, %1, off"
               :
               : "v"(lds_off), "v"(gptr)
               : "memory");
}

__device__ __forceinline__ void wait_async0() {
  asm volatile("s_wait_asynccnt 0x0" ::: "memory");
}

// ---------------------------------------------------------------------------
// Elementwise f32 -> bf16 (vectorized by 4)
// ---------------------------------------------------------------------------
__global__ __launch_bounds__(256) void cvt_bf16_kernel(
    const float* __restrict__ in, unsigned short* __restrict__ out, int n) {
  int i = (blockIdx.x * 256 + threadIdx.x) * 4;
  if (i >= n) return;
  float4 f = *(const float4*)(in + i);
  ushort4 s;
  s.x = f2bf(f.x); s.y = f2bf(f.y); s.z = f2bf(f.z); s.w = f2bf(f.w);
  *(ushort4*)(out + i) = s;
}

// ---------------------------------------------------------------------------
// Transpose-convert: W[K][N] f32  ->  Wt[N][K] bf16 (coalesced both sides)
// ---------------------------------------------------------------------------
__global__ __launch_bounds__(256) void cvt_transpose_kernel(
    const float* __restrict__ W, unsigned short* __restrict__ Wt, int K, int N) {
  __shared__ float tile[32][33];
  const int k0 = blockIdx.y * 32;
  const int n0 = blockIdx.x * 32;
  const int tx = threadIdx.x;   // 0..31
  const int ty = threadIdx.y;   // 0..7
#pragma unroll
  for (int i = 0; i < 4; ++i)
    tile[ty + 8 * i][tx] = W[(size_t)(k0 + ty + 8 * i) * N + n0 + tx];
  __syncthreads();
#pragma unroll
  for (int i = 0; i < 4; ++i)
    Wt[(size_t)(n0 + ty + 8 * i) * K + k0 + tx] = f2bf(tile[tx][ty + 8 * i]);
}

// ---------------------------------------------------------------------------
// bf16 WMMA GEMM:  C[M][N] = A[M][K] * Bt[N][K]^T + bias  (opt relu)
// 128 threads = 4 wave32; workgroup tile 64x64; K step 32.
// Statically double-buffered LDS, filled by global_load_async_to_lds_b128.
// ---------------------------------------------------------------------------
#define TM 64
#define TN 64
#define TK 32
#define LROW 40                       // bf16/row in LDS (80 B, 16B aligned)
#define BUFSTRIDE ((TM + TN) * LROW)  // ushorts per LDS buffer (10240 B)

// Issue the async copy of one 64x32 A-tile + 64x32 B-tile into LDS buffer lA.
__device__ __forceinline__ void issue_tile(
    const unsigned short* __restrict__ A, const unsigned short* __restrict__ Bt,
    unsigned short* lA, int m0, int n0, int K, int k0, int lr, int lc) {
  unsigned short* lB = lA + TM * LROW;
#pragma unroll
  for (int p = 0; p < 2; ++p) {
    const int row = p * 32 + lr;
    async_copy_b128(lds_off_of(lA + row * LROW + lc),
                    A + (size_t)(m0 + row) * K + k0 + lc);
    async_copy_b128(lds_off_of(lB + row * LROW + lc),
                    Bt + (size_t)(n0 + row) * K + k0 + lc);
  }
}

// Load all fragments first (10x ds_load_b128, staggered DScnt waits), then
// run the 4 WMMAs for this wave's 16x64 strip.
__device__ __forceinline__ void compute_tile(const unsigned short* cA,
                                             int wave, int lane, v8f acc[4]) {
  const unsigned short* cB = cA + TM * LROW;
  // --- A fragment (16x32 bf16, ISA 7.12.2 layout) ---
  // lanes 0-15: row m=lane,     K = 0..7   and 16..23
  // lanes 16-31: row m=lane-16, K = 8..15  and 24..31
  const int am  = lane & 15;
  const int akb = (lane < 16) ? 0 : 8;
  const unsigned short* ap = cA + (wave * 16 + am) * LROW;
  v16bf afrag;
  ((v4u*)&afrag)[0] = *(const v4u*)(ap + akb);        // 8 bf16
  ((v4u*)&afrag)[1] = *(const v4u*)(ap + akb + 16);   // 8 bf16

  // --- B fragments (32x16): lanes 0-15 cols w/ K 0..7,16..23; lanes 16-31 K 8..15,24..31
  const int bn  = lane & 15;
  const int bkb = (lane < 16) ? 0 : 8;
  v16bf bfrag[4];
#pragma unroll
  for (int t = 0; t < 4; ++t) {
    const unsigned short* bp = cB + (t * 16 + bn) * LROW + bkb;
    ((v4u*)&bfrag[t])[0] = *(const v4u*)(bp);
    ((v4u*)&bfrag[t])[1] = *(const v4u*)(bp + 16);
  }
#pragma unroll
  for (int t = 0; t < 4; ++t)
    acc[t] = __builtin_amdgcn_wmma_f32_16x16x32_bf16(
        false, afrag, false, bfrag[t], (short)0, acc[t], false, false);
}

__global__ __launch_bounds__(128) void gemm_bf16_wmma(
    const unsigned short* __restrict__ A,   // [M][K] bf16
    const unsigned short* __restrict__ Bt,  // [N][K] bf16 (pre-transposed B)
    const float* __restrict__ bias,         // [N]
    float* __restrict__ Cf,                 // optional f32 out
    unsigned short* __restrict__ Cb,        // optional bf16 out
    int M, int N, int K, int relu) {
  __shared__ __align__(16) unsigned short smem[2 * BUFSTRIDE];

  const int tid  = threadIdx.x;
  const int wave = tid >> 5;   // 0..3
  const int lane = tid & 31;   // wave32
  const int m0 = blockIdx.y * TM;
  const int n0 = blockIdx.x * TN;

  v8f acc[4] = {v8f{}, v8f{}, v8f{}, v8f{}};

  // cooperative async-copy indexing: 16B per thread per array per pass
  const int lr = tid >> 2;        // 0..31  (row within half-tile)
  const int lc = (tid & 3) * 8;   // bf16 column chunk

  const int nk = K / TK;  // 32 or 128 here: always even
  unsigned short* buf0 = smem;
  unsigned short* buf1 = smem + BUFSTRIDE;

  issue_tile(A, Bt, buf0, m0, n0, K, 0, lr, lc);  // prologue: tile 0 -> buf0

  for (int i = 0; i < nk; i += 2) {
    // ---- phase 0: consume buf0 (tile i), fill buf1 (tile i+1) ----
    wait_async0();    // this wave's copies into buf0 complete
    __syncthreads();  // all waves' copies complete; buf1 fully consumed
    if (i + 1 < nk)
      issue_tile(A, Bt, buf1, m0, n0, K, (i + 1) * TK, lr, lc);
    if (i + 2 < nk) {  // L2 prefetch two tiles ahead -> global_prefetch_b8
      const int kf = (i + 2) * TK;
      __builtin_prefetch(A + (size_t)(m0 + lr) * K + kf + lc, 0, 1);
      __builtin_prefetch(Bt + (size_t)(n0 + lr) * K + kf + lc, 0, 1);
    }
    compute_tile(buf0, wave, lane, acc);

    // ---- phase 1: consume buf1 (tile i+1), fill buf0 (tile i+2) ----
    wait_async0();
    __syncthreads();
    if (i + 2 < nk)
      issue_tile(A, Bt, buf0, m0, n0, K, (i + 2) * TK, lr, lc);
    if (i + 3 < nk) {
      const int kf = (i + 3) * TK;
      __builtin_prefetch(A + (size_t)(m0 + lr) * K + kf + lc, 0, 1);
      __builtin_prefetch(Bt + (size_t)(n0 + lr) * K + kf + lc, 0, 1);
    }
    compute_tile(buf1, wave, lane, acc);
    // no trailing barrier: each wave's LDS reads retire before it reaches the
    // next phase's barrier (s_wait_dscnt precedes the WMMAs), and writes to a
    // buffer are only issued after that barrier.
  }

  // --- epilogue: C/D layout: VGPR r -> M = r (lanes 0-15) / 8+r (lanes 16-31)
  const int cn    = lane & 15;
  const int mbase = m0 + wave * 16 + ((lane < 16) ? 0 : 8);
#pragma unroll
  for (int t = 0; t < 4; ++t) {
    const int n = n0 + t * 16 + cn;
    const float bn_ = bias[n];
#pragma unroll
    for (int r = 0; r < 8; ++r) {
      float v = acc[t][r] + bn_;
      if (relu) v = fmaxf(v, 0.0f);
      const size_t off = (size_t)(mbase + r) * N + n;
      if (Cf) Cf[off] = v;
      if (Cb) Cb[off] = f2bf(v);
    }
  }
}

// ---------------------------------------------------------------------------
// Fused residual + LayerNorm over rows of length 1024 (4 elems/thread).
// Emits f32 and (optionally) bf16 copies of the normalized row.
// ---------------------------------------------------------------------------
__global__ __launch_bounds__(256) void add_ln_kernel(
    const float* __restrict__ X, const float* __restrict__ Y,
    const float* __restrict__ g, const float* __restrict__ beta,
    float* __restrict__ outf, unsigned short* __restrict__ outb) {
  __shared__ float ssum[256];
  __shared__ float ssq[256];
  const int row = blockIdx.x;
  const float* x = X + (size_t)row * DMQ;
  const float* y = Y + (size_t)row * DMQ;

  float vals[4];
  float lsum = 0.f, lsq = 0.f;
#pragma unroll
  for (int i = 0; i < 4; ++i) {
    const int c = threadIdx.x + 256 * i;
    const float v = x[c] + y[c];
    vals[i] = v;
    lsum += v;
    lsq += v * v;
  }
  ssum[threadIdx.x] = lsum;
  ssq[threadIdx.x] = lsq;
  __syncthreads();
  for (int s = 128; s > 0; s >>= 1) {
    if (threadIdx.x < s) {
      ssum[threadIdx.x] += ssum[threadIdx.x + s];
      ssq[threadIdx.x] += ssq[threadIdx.x + s];
    }
    __syncthreads();
  }
  const float mu   = ssum[0] * (1.0f / DMQ);
  const float var  = ssq[0] * (1.0f / DMQ) - mu * mu;
  const float rstd = rsqrtf(var + 1e-5f);
#pragma unroll
  for (int i = 0; i < 4; ++i) {
    const int c = threadIdx.x + 256 * i;
    const float v = (vals[i] - mu) * rstd * g[c] + beta[c];
    const size_t off = (size_t)row * DMQ + c;
    outf[off] = v;
    if (outb) outb[off] = f2bf(v);
  }
}

// ---------------------------------------------------------------------------
extern "C" void kernel_launch(void* const* d_in, const int* in_sizes, int n_in,
                              void* d_out, int out_size, void* d_ws,
                              size_t ws_size, hipStream_t stream) {
  (void)in_sizes; (void)n_in; (void)out_size; (void)ws_size;
  const float* x   = (const float*)d_in[0];
  // d_in[1] = mask (unused: diagonal-only attention), d_in[2..5] = Wq,bq,Wk,bk (dead code)
  const float* Wv  = (const float*)d_in[6];
  const float* bv  = (const float*)d_in[7];
  const float* Wo  = (const float*)d_in[8];
  const float* bo  = (const float*)d_in[9];
  const float* W1  = (const float*)d_in[10];
  const float* b1  = (const float*)d_in[11];
  const float* W2  = (const float*)d_in[12];
  const float* b2  = (const float*)d_in[13];
  const float* g1  = (const float*)d_in[14];
  const float* be1 = (const float*)d_in[15];
  const float* g2  = (const float*)d_in[16];
  const float* be2 = (const float*)d_in[17];
  float* out = (float*)d_out;

  char* ws = (char*)d_ws;
  size_t o = 0;
  auto alloc = [&](size_t bytes) {
    size_t r = o;
    o += (bytes + 255) & ~(size_t)255;
    return r;
  };
  const size_t MB = 1024 * 1024;

  // Region A (32 MiB) holds xb|vb|att early, then is reused whole for ff1b.
  const size_t regionA = alloc(32 * MB);
  unsigned short* xb   = (unsigned short*)(ws + regionA);            // 8 MiB
  unsigned short* vb   = (unsigned short*)(ws + regionA + 8 * MB);   // 8 MiB
  float*          att  = (float*)(ws + regionA + 16 * MB);           // 16 MiB
  unsigned short* ff1b = (unsigned short*)(ws + regionA);            // 32 MiB reuse
  unsigned short* Wvt  = (unsigned short*)(ws + alloc(2 * MB));
  unsigned short* Wot  = (unsigned short*)(ws + alloc(2 * MB));
  unsigned short* W1t  = (unsigned short*)(ws + alloc(8 * MB));
  unsigned short* W2t  = (unsigned short*)(ws + alloc(8 * MB));
  float*          hf   = (float*)(ws + alloc(16 * MB));
  unsigned short* hb   = (unsigned short*)(ws + alloc(8 * MB));
  float*          ff2  = (float*)(ws + alloc(16 * MB));

  // --- one-time conversions ---
  cvt_bf16_kernel<<<dim3((ROWS * DMQ) / 1024), 256, 0, stream>>>(x, xb, ROWS * DMQ);
  cvt_transpose_kernel<<<dim3(DMQ / 32, DMQ / 32), dim3(32, 8), 0, stream>>>(Wv, Wvt, DMQ, DMQ);
  cvt_transpose_kernel<<<dim3(DMQ / 32, DMQ / 32), dim3(32, 8), 0, stream>>>(Wo, Wot, DMQ, DMQ);
  cvt_transpose_kernel<<<dim3(FFQ / 32, DMQ / 32), dim3(32, 8), 0, stream>>>(W1, W1t, DMQ, FFQ);
  cvt_transpose_kernel<<<dim3(DMQ / 32, FFQ / 32), dim3(32, 8), 0, stream>>>(W2, W2t, FFQ, DMQ);

  // --- v = x @ Wv + bv  (bf16 out) ---
  gemm_bf16_wmma<<<dim3(DMQ / TN, ROWS / TM), 128, 0, stream>>>(
      xb, Wvt, bv, nullptr, vb, ROWS, DMQ, DMQ, 0);
  // --- att = v @ Wo + bo  (f32 out) ---
  gemm_bf16_wmma<<<dim3(DMQ / TN, ROWS / TM), 128, 0, stream>>>(
      vb, Wot, bo, att, nullptr, ROWS, DMQ, DMQ, 0);
  // --- h = LN(x + att) (f32 + bf16) ---
  add_ln_kernel<<<dim3(ROWS), 256, 0, stream>>>(x, att, g1, be1, hf, hb);
  // --- ff1 = relu(h @ W1 + b1) (bf16 out; overwrites region A) ---
  gemm_bf16_wmma<<<dim3(FFQ / TN, ROWS / TM), 128, 0, stream>>>(
      hb, W1t, b1, nullptr, ff1b, ROWS, FFQ, DMQ, 1);
  // --- ff2 = ff1 @ W2 + b2 (f32 out) ---
  gemm_bf16_wmma<<<dim3(DMQ / TN, ROWS / TM), 128, 0, stream>>>(
      ff1b, W2t, b2, ff2, nullptr, ROWS, DMQ, FFQ, 0);
  // --- out = LN(h + ff2) ---
  add_ln_kernel<<<dim3(ROWS), 256, 0, stream>>>(hf, ff2, g2, be2, out, nullptr);
}
